// MHA_61675730371096
// MI455X (gfx1250) — matrix operand order repaired and checked by reading, not compiled, not run on hardware
//
#include <hip/hip_runtime.h>
#include <hip/hip_bf16.h>

// ---------------- problem constants ----------------
#define B_   16
#define M_   1024
#define HID_ 512
#define H_   8
#define PPAD 2
#define DH_  64
#define BH_  128          // B_*H_
#define LREAL 1026        // PPAD + 1024
#define LP   1088         // padded rows = 68 * 16 (both M and N sides)

typedef _Float16 v8h  __attribute__((ext_vector_type(8)));
typedef _Float16 v16h __attribute__((ext_vector_type(16)));
typedef float    v8f  __attribute__((ext_vector_type(8)));

// D = A(16x32 f16) * B(32x16 f16) + C(16x16 f32)
__device__ __forceinline__ v8f wmma_f16(v16h a, v16h b, v8f c) {
    return __builtin_amdgcn_wmma_f32_16x16x32_f16(false, a, false, b,
                                                  (short)0, c, false, false);
}

// A-operand gather for one lane: row ptr = Q + m*DH_, kb = (lane>>4)*8,
// dbase in {0,32}. Lane layout (contiguous k order, matches B):
//   halves 0..7  -> K = dbase + 2*kb + i
//   halves 8..15 -> K = dbase + 2*kb + 8 + (i-8)
__device__ __forceinline__ v16h load_a(const _Float16* row, int kb, int dbase) {
    v8h lo = *(const v8h*)(row + dbase + 2 * kb);
    v8h hi = *(const v8h*)(row + dbase + 2 * kb + 8);
    return __builtin_shufflevector(lo, hi, 0,1,2,3,4,5,6,7,8,9,10,11,12,13,14,15);
}

// ---------------- kernel 1: pad + split heads + f16 convert + V transpose ----
__global__ void xattn_prep(const float* __restrict__ X, const float* __restrict__ Y,
                           const float* __restrict__ padx, const float* __restrict__ pady,
                           _Float16* __restrict__ Qx, _Float16* __restrict__ Qy,
                           _Float16* __restrict__ Vtx, _Float16* __restrict__ Vty) {
    size_t tid = (size_t)blockIdx.x * blockDim.x + threadIdx.x;
    const size_t total = (size_t)BH_ * LP * DH_;
    if (tid >= total) return;
    int d  = (int)(tid & (DH_ - 1));
    int m  = (int)((tid >> 6) % LP);
    int bh = (int)(tid / ((size_t)LP * DH_));
    int b = bh >> 3, h = bh & 7;
    int col = h * DH_ + d;
    float vx = 0.f, vy = 0.f;
    if (m < PPAD) {
        vx = padx[m * HID_ + col];
        vy = pady[m * HID_ + col];
    } else if (m < LREAL) {
        size_t off = ((size_t)b * M_ + (m - PPAD)) * HID_ + col;
        vx = X[off]; vy = Y[off];
    }
    _Float16 hx = (_Float16)vx, hy = (_Float16)vy;
    size_t qoff = (size_t)bh * LP * DH_ + (size_t)m * DH_ + d;
    Qx[qoff] = hx; Qy[qoff] = hy;
    size_t toff = (size_t)bh * DH_ * LP + (size_t)d * LP + m;
    Vtx[toff] = hx; Vty[toff] = hy;
}

// ---------------- kernel 2: row softmax stats of Q*K^T (max & sumexp) -------
// grid (LP/64, BH_), block 128 (4 waves); one wave per 16-row tile.
__global__ void xattn_stats(const _Float16* __restrict__ Q, const _Float16* __restrict__ K,
                            float* __restrict__ omax, float* __restrict__ osum) {
    int lane = threadIdx.x & 31;
    int wid  = threadIdx.x >> 5;
    int bh   = blockIdx.y;
    int mbase = (blockIdx.x * 4 + wid) * 16;
    int lo16 = lane & 15, hi2 = lane >> 4;
    int kb = hi2 * 8, khalf = hi2 * 16;

    const _Float16* Qb = Q + (size_t)bh * LP * DH_;
    const _Float16* Kb = K + (size_t)bh * LP * DH_;
    const _Float16* qrow = Qb + (size_t)(mbase + lo16) * DH_;
    v16h a0 = load_a(qrow, kb, 0);
    v16h a1 = load_a(qrow, kb, 32);

    float lm[8], ls[8];
#pragma unroll
    for (int j = 0; j < 8; ++j) { lm[j] = -3.0e38f; ls[j] = 0.f; }

    for (int nb = 0; nb < LP; nb += 16) {
        int n = nb + lo16;
        const _Float16* krow = Kb + (size_t)n * DH_;
        v16h b0 = *(const v16h*)(krow + khalf);
        v16h b1 = *(const v16h*)(krow + 32 + khalf);
        v8f s = {};
        s = wmma_f16(a0, b0, s);
        s = wmma_f16(a1, b1, s);
        bool valid = n < LREAL;
#pragma unroll
        for (int j = 0; j < 8; ++j) {
            float sv = valid ? s[j] : -3.0e38f;
            float nm = fmaxf(lm[j], sv);
            ls[j] = ls[j] * __expf(lm[j] - nm) + __expf(sv - nm);
            lm[j] = nm;
        }
    }
    // merge across the 16 lanes sharing a row group (low 4 lane bits)
#pragma unroll
    for (int j = 0; j < 8; ++j) {
#pragma unroll
        for (int st = 1; st <= 8; st <<= 1) {
            float om = __shfl_xor(lm[j], st, 32);
            float os = __shfl_xor(ls[j], st, 32);
            float nm = fmaxf(lm[j], om);
            ls[j] = ls[j] * __expf(lm[j] - nm) + os * __expf(om - nm);
            lm[j] = nm;
        }
    }
    if (lo16 == 0) {
        int rbase = bh * LP + mbase + hi2 * 8;
#pragma unroll
        for (int j = 0; j < 8; ++j) { omax[rbase + j] = lm[j]; osum[rbase + j] = ls[j]; }
    }
}

// ---------------- kernel 3: O = softmax_row(Q*K^T) * V ----------------------
// Vt is pre-transposed [bh][d][n]. grid (LP/64, BH_), block 128.
__global__ void xattn_apply(const _Float16* __restrict__ Q, const _Float16* __restrict__ K,
                            const _Float16* __restrict__ Vt,
                            const float* __restrict__ rmax, const float* __restrict__ rsum,
                            float* __restrict__ out) {
    __shared__ __align__(32) _Float16 Pbuf[4][16 * 32];
    int lane = threadIdx.x & 31;
    int wid  = threadIdx.x >> 5;
    int bh   = blockIdx.y;
    int b = bh >> 3, h = bh & 7;
    int mbase = (blockIdx.x * 4 + wid) * 16;
    int lo16 = lane & 15, hi2 = lane >> 4;
    int kb = hi2 * 8, khalf = hi2 * 16;

    const _Float16* Qb = Q  + (size_t)bh * LP * DH_;
    const _Float16* Kb = K  + (size_t)bh * LP * DH_;
    const _Float16* Vb = Vt + (size_t)bh * DH_ * LP;
    const _Float16* qrow = Qb + (size_t)(mbase + lo16) * DH_;
    v16h a0 = load_a(qrow, kb, 0);
    v16h a1 = load_a(qrow, kb, 32);

    float rm[8], rinv[8];
    {
        int rbase = bh * LP + mbase + hi2 * 8;
#pragma unroll
        for (int j = 0; j < 8; ++j) {
            rm[j]   = rmax[rbase + j];
            rinv[j] = 1.0f / rsum[rbase + j];
        }
    }

    v8f acc0 = {}, acc1 = {}, acc2 = {}, acc3 = {};
    _Float16* pb = Pbuf[wid];

    for (int nb = 0; nb < LP; nb += 32) {
        // two 16-wide S sub-tiles -> normalized P in LDS (C-layout -> A-layout bounce)
#pragma unroll
        for (int t = 0; t < 2; ++t) {
            int n = nb + t * 16 + lo16;
            const _Float16* krow = Kb + (size_t)n * DH_;
            v16h b0 = *(const v16h*)(krow + khalf);
            v16h b1 = *(const v16h*)(krow + 32 + khalf);
            v8f s = {};
            s = wmma_f16(a0, b0, s);
            s = wmma_f16(a1, b1, s);
            bool valid = n < LREAL;
#pragma unroll
            for (int j = 0; j < 8; ++j) {
                float p = valid ? __expf(s[j] - rm[j]) * rinv[j] : 0.f;
                pb[(j + 8 * hi2) * 32 + t * 16 + lo16] = (_Float16)p;
            }
        }
        // read P back as an A operand (per-wave LDS slice; in-wave DS ordering)
        v8h plo = *(const v8h*)(pb + lo16 * 32 + 2 * kb);
        v8h phi = *(const v8h*)(pb + lo16 * 32 + 2 * kb + 8);
        v16h pa = __builtin_shufflevector(plo, phi, 0,1,2,3,4,5,6,7,8,9,10,11,12,13,14,15);
        // O += P * V over four 16-wide d chunks (B operand contiguous in Vt)
        {
            const _Float16* v0 = Vb + (size_t)(0 * 16 + lo16) * LP + nb + khalf;
            const _Float16* v1 = Vb + (size_t)(1 * 16 + lo16) * LP + nb + khalf;
            const _Float16* v2 = Vb + (size_t)(2 * 16 + lo16) * LP + nb + khalf;
            const _Float16* v3 = Vb + (size_t)(3 * 16 + lo16) * LP + nb + khalf;
            acc0 = wmma_f16(pa, *(const v16h*)v0, acc0);
            acc1 = wmma_f16(pa, *(const v16h*)v1, acc1);
            acc2 = wmma_f16(pa, *(const v16h*)v2, acc2);
            acc3 = wmma_f16(pa, *(const v16h*)v3, acc3);
        }
    }

#pragma unroll
    for (int j = 0; j < 8; ++j) {
        int m = mbase + 8 * hi2 + j;
        if (m >= PPAD && m < LREAL) {
            size_t o = ((size_t)b * M_ + (m - PPAD)) * HID_ + h * DH_ + lo16;
            out[o + 0]  = acc0[j];
            out[o + 16] = acc1[j];
            out[o + 32] = acc2[j];
            out[o + 48] = acc3[j];
        }
    }
}

// ---------------- launcher --------------------------------------------------
extern "C" void kernel_launch(void* const* d_in, const int* in_sizes, int n_in,
                              void* d_out, int out_size, void* d_ws, size_t ws_size,
                              hipStream_t stream) {
    const float* X    = (const float*)d_in[0];
    const float* Y    = (const float*)d_in[1];
    const float* padx = (const float*)d_in[2];
    const float* pady = (const float*)d_in[3];
    float* out = (float*)d_out;

    char* ws = (char*)d_ws;
    const size_t qbytes = (size_t)BH_ * LP * DH_ * sizeof(_Float16); // 17,825,792 B
    _Float16* Qx  = (_Float16*)(ws + 0 * qbytes);
    _Float16* Qy  = (_Float16*)(ws + 1 * qbytes);
    _Float16* Vtx = (_Float16*)(ws + 2 * qbytes);
    _Float16* Vty = (_Float16*)(ws + 3 * qbytes);
    float* stat = (float*)(ws + 4 * qbytes);
    const size_t srows = (size_t)BH_ * LP;
    float* mx = stat;              // row softmax stats for Q=Qx (softmax over n)
    float* sx = mx + srows;
    float* my = sx + srows;        // row softmax stats for Q=Qy (== column softmax of sim)
    float* sy = my + srows;

    const size_t total = (size_t)BH_ * LP * DH_;
    xattn_prep<<<dim3((unsigned)((total + 255) / 256)), dim3(256), 0, stream>>>(
        X, Y, padx, pady, Qx, Qy, Vtx, Vty);

    dim3 grid(LP / 64, BH_);   // 17 x 128, 4 waves per block, one 16-row tile per wave
    dim3 block(128);
    xattn_stats<<<grid, block, 0, stream>>>(Qx, Qy, mx, sx);
    xattn_stats<<<grid, block, 0, stream>>>(Qy, Qx, my, sy);

    // X_in_Y = softmax over m of sim, output indexed by n  ->  Q=Yp, K=Xp, V=Xp
    xattn_apply<<<grid, block, 0, stream>>>(Qy, Qx, Vtx, my, sy, out);
    // Y_in_X = softmax over n of sim, output indexed by m  ->  Q=Xp, K=Yp, V=Yp
    xattn_apply<<<grid, block, 0, stream>>>(Qx, Qy, Vty, mx, sx,
                                            out + (size_t)B_ * M_ * HID_);
}
